// CRFModel_50448685858818
// MI455X (gfx1250) — matrix operand order repaired and checked by reading, not compiled, not run on hardware
//
#include <hip/hip_runtime.h>
#include <math.h>

typedef __attribute__((ext_vector_type(16))) _Float16 v16h;
typedef __attribute__((ext_vector_type(8)))  float    v8f;

constexpr int E = 128, C = 16, B = 64, T = 4096;
constexpr int ROWS  = B * T;        // 262144 (b,t) rows
constexpr int TILES = ROWS / 16;    // 16384 WMMA row-tiles

// ---------------------------------------------------------------------------
// Kernel 1: emissions  em[row, c] = sum_e embedding[x[row], e] * fc_w[e, c] + fc_b[c]
// One wave per 16-row tile iteration; 4x v_wmma_f32_16x16x32_f16 per tile (K=128).
// B-fragments (fc_w) are built once per wave and reused across its tiles.
// ---------------------------------------------------------------------------
__global__ __launch_bounds__(256) void emissions_wmma(
    const int* __restrict__ x, const float* __restrict__ embedding,
    const float* __restrict__ fc_w, const float* __restrict__ fc_b,
    float* __restrict__ em)
{
  const int lane  = threadIdx.x & 31;
  const int wIn   = threadIdx.x >> 5;
  const int wPerB = blockDim.x >> 5;
  const int wave  = blockIdx.x * wPerB + wIn;
  const int nWave = gridDim.x * wPerB;

  const int n = lane & 15;   // class column for B / C / D fragments
  const int h = lane >> 4;   // half selector

  // B fragment: fc_w is (E, C) row-major. VGPR j of ktile kt holds
  // K = 32*kt + 16*h + 2*j (+1), column n.
  v16h bB[4];
#pragma unroll
  for (int kt = 0; kt < 4; ++kt) {
#pragma unroll
    for (int j = 0; j < 8; ++j) {
      const int e0 = 32 * kt + 16 * h + 2 * j;
      bB[kt][2 * j]     = (_Float16)fc_w[e0 * C + n];
      bB[kt][2 * j + 1] = (_Float16)fc_w[(e0 + 1) * C + n];
    }
  }
  const float bias = fc_b[n];

  for (int tile = wave; tile < TILES; tile += nWave) {
    const int rowBase = tile * 16;
    const int m   = lane & 15;                 // A row within tile
    const int tok = x[rowBase + m];
    const float* rowp = embedding + (long long)tok * E;

    v8f acc = {};
#pragma unroll
    for (int kt = 0; kt < 4; ++kt) {
      // A fragment: VGPR j holds K = 16*h + 2*j (+1) of ktile kt.
      v16h a;
#pragma unroll
      for (int j = 0; j < 8; ++j) {
        const int e0 = 32 * kt + 16 * h + 2 * j;
        const float2 p = *(const float2*)(rowp + e0);
        a[2 * j]     = (_Float16)p.x;
        a[2 * j + 1] = (_Float16)p.y;
      }
      acc = __builtin_amdgcn_wmma_f32_16x16x32_f16(
          /*neg_a=*/false, a, /*neg_b=*/false, bB[kt],
          /*c_mod=*/(short)0, acc, /*reuse_a=*/false, /*reuse_b=*/false);
    }
    // D: VGPR r -> row 8*h + r, column n
#pragma unroll
    for (int r = 0; r < 8; ++r) {
      const int row = rowBase + 8 * h + r;
      em[row * C + n] = acc[r] + bias;
    }
  }
}

// ---------------------------------------------------------------------------
// Kernel 2: per-batch CRF score. One wave per batch.
//   Prepass (32 lanes): gold-path numerator + sequence length (mask is a prefix).
//   Scan (lane = next-class, duplicated in upper 16 lanes): forward algorithm.
// Writes (logZ - num) per batch into part[].
// ---------------------------------------------------------------------------
__global__ __launch_bounds__(32) void crf_scan(
    const int* __restrict__ x, const int* __restrict__ tags,
    const float* __restrict__ em,
    const float* __restrict__ start_tr, const float* __restrict__ end_tr,
    const float* __restrict__ trans, float* __restrict__ part)
{
  const int b    = blockIdx.x;
  const int lane = threadIdx.x;
  const int n    = lane & 15;

  const int*   xb  = x    + b * T;
  const int*   tb  = tags + b * T;
  const float* emb = em   + (long long)b * T * C;

  // ---- numerator + length ----
  float num = 0.f;
  int   cnt = 0;
  for (int t = lane; t < T; t += 32) {
    const int xt = xb[t];
    if (xt != 0) {
      ++cnt;
      const int tg = tb[t];
      num += emb[t * C + tg];
      if (t > 0) num += trans[tb[t - 1] * C + tg];
    }
  }
#pragma unroll
  for (int off = 16; off; off >>= 1) {
    num += __shfl_xor(num, off, 32);
    cnt += __shfl_xor(cnt, off, 32);
  }
  const int len = cnt;                       // mask is a length-prefix
  num += start_tr[tb[0]] + end_tr[tb[len - 1]];

  // ---- forward algorithm ----
  float tcol[16];                            // trans[c -> n], resident in VGPRs
#pragma unroll
  for (int c = 0; c < 16; ++c) tcol[c] = trans[c * C + n];

  float alpha = start_tr[n] + emb[n];        // t = 0

  for (int t = 1; t < len; ++t) {
    float v[16];
#pragma unroll
    for (int c = 0; c < 16; ++c) v[c] = __shfl(alpha, c, 16) + tcol[c];
    float mx = v[0];
#pragma unroll
    for (int c = 1; c < 16; ++c) mx = fmaxf(mx, v[c]);
    float s = 0.f;
#pragma unroll
    for (int c = 0; c < 16; ++c) s += __expf(v[c] - mx);
    alpha = emb[t * C + n] + mx + __logf(s);
  }

  // logZ = logsumexp_n(alpha + end_tr[n])
  float a  = alpha + end_tr[n];
  float mx = a;
#pragma unroll
  for (int off = 8; off; off >>= 1) mx = fmaxf(mx, __shfl_xor(mx, off, 16));
  float s = __expf(a - mx);
#pragma unroll
  for (int off = 8; off; off >>= 1) s += __shfl_xor(s, off, 16);
  const float logZ = mx + __logf(s);

  if (lane == 0) part[b] = logZ - num;       // contribution to -llh
}

// ---------------------------------------------------------------------------
// Kernel 3: deterministic final reduction over B partials.
// ---------------------------------------------------------------------------
__global__ void final_reduce(const float* __restrict__ part, float* __restrict__ out)
{
  float s = 0.f;
  for (int i = 0; i < B; ++i) s += part[i];
  out[0] = s;
}

extern "C" void kernel_launch(void* const* d_in, const int* in_sizes, int n_in,
                              void* d_out, int out_size, void* d_ws, size_t ws_size,
                              hipStream_t stream)
{
  const int*   x         = (const int*)d_in[0];
  const int*   tags      = (const int*)d_in[1];
  const float* embedding = (const float*)d_in[2];
  const float* fc_w      = (const float*)d_in[3];
  const float* fc_b      = (const float*)d_in[4];
  const float* start_tr  = (const float*)d_in[5];
  const float* end_tr    = (const float*)d_in[6];
  const float* trans     = (const float*)d_in[7];
  float* out = (float*)d_out;

  float* em   = (float*)d_ws;                    // ROWS * C floats = 16 MiB
  float* part = em + (size_t)ROWS * C;           // B floats

  emissions_wmma<<<256, 256, 0, stream>>>(x, embedding, fc_w, fc_b, em);
  crf_scan<<<B, 32, 0, stream>>>(x, tags, em, start_tr, end_tr, trans, part);
  final_reduce<<<1, 1, 0, stream>>>(part, out);
}
